// SparseMAEDecoder_17849884082701
// MI455X (gfx1250) — compile-verified, matrix operand order checked by reading, not run on hardware
//
#include <hip/hip_runtime.h>

// ---------------------------------------------------------------------------
// CDNA5 (gfx1250) sparse MAE decoder: implicit-GEMM conv via v_wmma_f32_16x16x32_f16
// wave32; each wave computes a 16(pixels) x 64(cout) tile = 4 WMMA n-tiles.
// Subm layers: per-tap B fragments staged in LDS (block-wide reuse, ds_load path).
// ---------------------------------------------------------------------------

typedef __attribute__((ext_vector_type(16))) _Float16 v16h;
typedef __attribute__((ext_vector_type(8)))  _Float16 v8h;
typedef __attribute__((ext_vector_type(8)))  float    v8f;
typedef _Float16 half_t;

union AFrag {
    v16h v;
    struct Halves { v8h lo; v8h hi; } p;
};

// ---------------------------------------------------------------------------
// Weight packing: W[3][3][CIN][COUT] f32 (HWIO)  ->  f16 B-fragments.
// Fragment layout (per CDNA5 ISA 16-bit 32x16 B matrix, wave32):
//   frag[t][kc][nt][lane][e], lane holds column n = lane&15,
//   k(e,lane) = (e<8?0:16) + (lane>=16?8:0) + (e&7), relative to kc*32.
// Flat: (((t*KC + kc)*NT + nt)*32 + lane)*16 + e
// ---------------------------------------------------------------------------
__global__ void pack_weights(const float* __restrict__ W, half_t* __restrict__ out,
                             int CIN, int COUT) {
    const int total = 9 * CIN * COUT;
    const int i = blockIdx.x * 256 + threadIdx.x;
    if (i >= total) return;
    const int NT = COUT >> 4, KC = CIN >> 5;
    const int e    = i & 15;
    const int lane = (i >> 4) & 31;
    int rest = i >> 9;
    const int nt = rest % NT; rest /= NT;
    const int kc = rest % KC;
    const int t  = rest / KC;
    const int k    = ((e < 8) ? 0 : 16) + ((lane >= 16) ? 8 : 0) + (e & 7);
    const int cin  = kc * 32 + k;
    const int cout = nt * 16 + (lane & 15);
    out[i] = (half_t)W[((long long)t * CIN + cin) * COUT + cout];
}

// Fold BN (+ optional conv bias) into y*scale + shift
__global__ void bn_prep(const float* __restrict__ g, const float* __restrict__ be,
                        const float* __restrict__ mu, const float* __restrict__ v,
                        const float* __restrict__ bias,
                        float* __restrict__ scale, float* __restrict__ shift, int C) {
    const int c = blockIdx.x * 256 + threadIdx.x;
    if (c >= C) return;
    const float s  = g[c] * rsqrtf(v[c] + 1e-5f);
    const float b0 = bias ? bias[c] : 0.0f;
    scale[c] = s;
    shift[c] = be[c] + (b0 - mu[c]) * s;
}

// x * m3 -> f16
__global__ void mask_cast(const float* __restrict__ x, const unsigned char* __restrict__ m,
                          half_t* __restrict__ out, int C, long long total) {
    const long long i = (long long)blockIdx.x * 256 + threadIdx.x;
    if (i >= total) return;
    const float val = m[i / C] ? x[i] : 0.0f;
    out[i] = (half_t)val;
}

// ---------------------------------------------------------------------------
// Inner K loop with B read from global (used by inverse-conv kernels).
// A fragment per lane: row m = lane&15 (pixel), two contiguous 8-channel groups
// at kc*32 + (lane>=16?8:0) and +16 more -> two b128 loads.
// ---------------------------------------------------------------------------
template <int CIN, int COUT>
__device__ __forceinline__ void mma_taps(const half_t* __restrict__ arow,
                                         const half_t* __restrict__ brow,
                                         bool ok, v8f (&acc)[4]) {
    constexpr int KC = CIN / 32, NT = COUT / 16;
#pragma unroll 4
    for (int kc = 0; kc < KC; ++kc) {
        AFrag a;
        { v16h z = {}; a.v = z; }
        if (ok) {
            a.p.lo = *(const v8h*)(arow + kc * 32);
            a.p.hi = *(const v8h*)(arow + kc * 32 + 16);
            if (kc + 1 < KC) __builtin_prefetch(arow + (kc + 1) * 32, 0, 1);
        }
        const half_t* bb = brow + (long long)kc * (NT * 512);
#pragma unroll
        for (int nt = 0; nt < 4; ++nt) {
            v16h bf = *(const v16h*)(bb + nt * 512);
            acc[nt] = __builtin_amdgcn_wmma_f32_16x16x32_f16(
                false, a.v, false, bf, (short)0, acc[nt], false, false);
        }
    }
}

// ---------------------------------------------------------------------------
// Submanifold 3x3 conv, pad 1, stride 1 + BN + ReLU + output-site mask.
// Block = 8 waves sharing (b, y, ct); wave w covers x-tile (xg*8 + w).
// Per tap: block stages B fragments (KC x 4nt x 512 halfs) into LDS, then
// every wave runs the K loop with ds_load B + global b128 A.
// ---------------------------------------------------------------------------
template <int CIN, int COUT>
__global__ void __launch_bounds__(256, 1)
subm_conv_bn_relu(const half_t* __restrict__ in, const half_t* __restrict__ wf,
                  const float* __restrict__ scale, const float* __restrict__ shift,
                  const unsigned char* __restrict__ mask,
                  half_t* __restrict__ out16, float* __restrict__ out32,
                  int B, int H, int W) {
    constexpr int KC = CIN / 32, NT = COUT / 16, CT = COUT / 64;
    __shared__ half_t ldsB[KC * 2048];  // one tap: KC chunks x (4 ntiles x 512 halfs)

    const int tid  = threadIdx.x;
    const int lane = tid & 31;
    const int wave = tid >> 5;
    const int WT = W >> 4;
    const int XG = WT >> 3;             // groups of 8 x-tiles per row (>=1 for all layers)
    int bid = blockIdx.x;
    const int ct = bid % CT; bid /= CT;
    const int xg = bid % XG; bid /= XG;
    const int y  = bid % H;
    const int b  = bid / H;
    const int x0 = ((xg << 3) + wave) << 4;
    const int m  = lane & 15;
    const int hi = lane >> 4;

    v8f acc[4];
    { v8f z = {}; acc[0] = z; acc[1] = z; acc[2] = z; acc[3] = z; }

    const int px = x0 + m;

    for (int ky = 0; ky < 3; ++ky) {
        const int iy = y + ky - 1;
        const bool rowOK = (iy >= 0) && (iy < H);     // block-uniform
        for (int kx = 0; kx < 3; ++kx) {
            const int t = ky * 3 + kx;
            __syncthreads();                          // previous tap fully consumed
            if (rowOK) {
                // cooperative stage: 2048 halfs per kc chunk = 256 threads x v8h
#pragma unroll
                for (int kc = 0; kc < KC; ++kc) {
                    const half_t* src =
                        wf + (long long)((t * KC + kc) * NT + ct * 4) * 512;
                    *(v8h*)(ldsB + kc * 2048 + tid * 8) = *(const v8h*)(src + tid * 8);
                }
            }
            __syncthreads();
            if (!rowOK) continue;                     // block-uniform (barriers done)

            const int ix = px + kx - 1;
            const bool ok = (ix >= 0) && (ix < W);    // per-lane edge handling
            const half_t* arow =
                in + (long long)((b * H + iy) * W + ix) * CIN + hi * 8;
#pragma unroll 4
            for (int kc = 0; kc < KC; ++kc) {
                AFrag a;
                { v16h z = {}; a.v = z; }
                if (ok) {
                    a.p.lo = *(const v8h*)(arow + kc * 32);
                    a.p.hi = *(const v8h*)(arow + kc * 32 + 16);
                    if (kc + 1 < KC) __builtin_prefetch(arow + (kc + 1) * 32, 0, 1);
                }
                const half_t* bb = ldsB + kc * 2048 + lane * 16;
#pragma unroll
                for (int nt = 0; nt < 4; ++nt) {
                    v16h bf = *(const v16h*)(bb + nt * 512);
                    acc[nt] = __builtin_amdgcn_wmma_f32_16x16x32_f16(
                        false, a.v, false, bf, (short)0, acc[nt], false, false);
                }
            }
        }
    }

    float sc[4], sh[4];
#pragma unroll
    for (int nt = 0; nt < 4; ++nt) {
        const int c = ct * 64 + nt * 16 + m;
        sc[nt] = scale[c]; sh[nt] = shift[c];
    }
#pragma unroll
    for (int r = 0; r < 8; ++r) {
        const int mo  = r + hi * 8;                   // D layout: m = vgpr + 8*(lane>=16)
        const int pix = (b * H + y) * W + x0 + mo;
        const float mk = mask[pix] ? 1.0f : 0.0f;
#pragma unroll
        for (int nt = 0; nt < 4; ++nt) {
            float v = fmaxf(acc[nt][r] * sc[nt] + sh[nt], 0.0f) * mk;
            const long long addr = (long long)pix * COUT + ct * 64 + nt * 16 + m;
            if (out32) out32[addr] = v;
            else       out16[addr] = (half_t)v;
        }
    }
}

// ---------------------------------------------------------------------------
// Inverse conv (transposed, lhs_dilation=2, pad (1,2), out = 2*in) + BN+ReLU+mask.
// Per output parity: even coord -> 1 tap (W row/col 1) at i = o/2;
// odd -> taps (W 0) at i = o>>1 and (W 2) at i = (o>>1)+1 (if in range).
// One wave = 16 same-parity outputs of one output row x 64 couts.
// Tap set depends on per-wave parity -> stays wave-independent (B via L2).
// ---------------------------------------------------------------------------
template <int CIN, int COUT>
__global__ void __launch_bounds__(256, 1)
inv_conv_bn_relu(const half_t* __restrict__ in, const half_t* __restrict__ wf,
                 const float* __restrict__ scale, const float* __restrict__ shift,
                 const unsigned char* __restrict__ mask,
                 half_t* __restrict__ out16,
                 int B, int H, int W) {  // input dims; output is 2H x 2W
    constexpr int KC = CIN / 32, NT = COUT / 16, CT = COUT / 64;
    const int Ho = H << 1, Wo = W << 1;
    const int lane = threadIdx.x & 31;
    const int wave = threadIdx.x >> 5;
    const int tile = blockIdx.x * 8 + wave;
    const int JT = W >> 4;
    const int ct = tile % CT;
    int pt = tile / CT;
    if (pt >= B * Ho * 2 * JT) return;
    const int jt  = pt % JT; pt /= JT;
    const int pxp = pt & 1;  pt >>= 1;
    const int oy  = pt % Ho;
    const int b   = pt / Ho;
    const int j0  = jt << 4;
    const int m   = lane & 15;
    const int hi  = lane >> 4;

    v8f acc[4];
    { v8f z = {}; acc[0] = z; acc[1] = z; acc[2] = z; acc[3] = z; }

    const int j = j0 + m;  // input column for this lane's A row

    int ny, iy0, wy0, iy1, wy1;
    if ((oy & 1) == 0) { ny = 1; iy0 = oy >> 1; wy0 = 1; iy1 = 0; wy1 = 0; }
    else {
        iy0 = oy >> 1;  wy0 = 0;
        iy1 = iy0 + 1;  wy1 = 2;
        ny  = (iy1 < H) ? 2 : 1;
    }
    const int nx = pxp ? 2 : 1;

    for (int ty = 0; ty < ny; ++ty) {
        const int iy = ty ? iy1 : iy0;
        const int wy = ty ? wy1 : wy0;
        for (int tx = 0; tx < nx; ++tx) {
            const int wx = pxp ? (tx ? 2 : 0) : 1;
            const int dx = pxp ? tx : 0;
            const int ix = j + dx;
            const bool ok = ix < W;            // per-lane right-edge handling
            const int t = wy * 3 + wx;
            const half_t* arow = in + (long long)((b * H + iy) * W + ix) * CIN + hi * 8;
            const half_t* brow = wf + (long long)((t * KC) * NT + ct * 4) * 512 + lane * 16;
            mma_taps<CIN, COUT>(arow, brow, ok, acc);
        }
    }

    float sc[4], sh[4];
#pragma unroll
    for (int nt = 0; nt < 4; ++nt) {
        const int c = ct * 64 + nt * 16 + m;
        sc[nt] = scale[c]; sh[nt] = shift[c];
    }
#pragma unroll
    for (int r = 0; r < 8; ++r) {
        const int mo  = r + hi * 8;
        const int ox  = ((j0 + mo) << 1) + pxp;
        const int pix = (b * Ho + oy) * Wo + ox;
        const float mk = mask[pix] ? 1.0f : 0.0f;
#pragma unroll
        for (int nt = 0; nt < 4; ++nt) {
            float v = fmaxf(acc[nt][r] * sc[nt] + sh[nt], 0.0f) * mk;
            out16[(long long)pix * COUT + ct * 64 + nt * 16 + m] = (half_t)v;
        }
    }
}

// ---------------------------------------------------------------------------
// Host: pipeline orchestration (all on `stream`, no allocation, deterministic)
// ---------------------------------------------------------------------------
extern "C" void kernel_launch(void* const* d_in, const int* in_sizes, int n_in,
                              void* d_out, int out_size, void* d_ws, size_t ws_size,
                              hipStream_t stream) {
    (void)in_sizes; (void)n_in; (void)out_size; (void)ws_size;
    int i = 0;
    const float* x            = (const float*)d_in[i++];
    const unsigned char* m3   = (const unsigned char*)d_in[i++];
    const unsigned char* m2   = (const unsigned char*)d_in[i++];
    const unsigned char* m1   = (const unsigned char*)d_in[i++];
    const unsigned char* m0   = (const unsigned char*)d_in[i++];
    const float *W_up1 = (const float*)d_in[i++], *b_up1 = (const float*)d_in[i++],
                *g_up1 = (const float*)d_in[i++], *be_up1 = (const float*)d_in[i++],
                *mu_up1 = (const float*)d_in[i++], *v_up1 = (const float*)d_in[i++];
    const float *W_dec1 = (const float*)d_in[i++], *g_dec1 = (const float*)d_in[i++],
                *be_dec1 = (const float*)d_in[i++], *mu_dec1 = (const float*)d_in[i++],
                *v_dec1 = (const float*)d_in[i++];
    const float *W_up2 = (const float*)d_in[i++], *b_up2 = (const float*)d_in[i++],
                *g_up2 = (const float*)d_in[i++], *be_up2 = (const float*)d_in[i++],
                *mu_up2 = (const float*)d_in[i++], *v_up2 = (const float*)d_in[i++];
    const float *W_dec2 = (const float*)d_in[i++], *g_dec2 = (const float*)d_in[i++],
                *be_dec2 = (const float*)d_in[i++], *mu_dec2 = (const float*)d_in[i++],
                *v_dec2 = (const float*)d_in[i++];
    const float *W_up3 = (const float*)d_in[i++], *b_up3 = (const float*)d_in[i++],
                *g_up3 = (const float*)d_in[i++], *be_up3 = (const float*)d_in[i++],
                *mu_up3 = (const float*)d_in[i++], *v_up3 = (const float*)d_in[i++];
    const float *W_dec3 = (const float*)d_in[i++], *g_dec3 = (const float*)d_in[i++],
                *be_dec3 = (const float*)d_in[i++], *mu_dec3 = (const float*)d_in[i++],
                *v_dec3 = (const float*)d_in[i++];

    char* ws = (char*)d_ws;
    size_t off = 0;
    auto take = [&](size_t bytes) -> void* {
        void* p = ws + off;
        off = (off + bytes + 255) & ~((size_t)255);
        return p;
    };
    half_t* wfU1 = (half_t*)take(sizeof(half_t) * 9 * 512 * 256);
    half_t* wfD1 = (half_t*)take(sizeof(half_t) * 9 * 256 * 256);
    half_t* wfU2 = (half_t*)take(sizeof(half_t) * 9 * 256 * 128);
    half_t* wfD2 = (half_t*)take(sizeof(half_t) * 9 * 128 * 128);
    half_t* wfU3 = (half_t*)take(sizeof(half_t) * 9 * 128 * 64);
    half_t* wfD3 = (half_t*)take(sizeof(half_t) * 9 * 64 * 64);
    float *scU1 = (float*)take(256 * 4), *shU1 = (float*)take(256 * 4);
    float *scD1 = (float*)take(256 * 4), *shD1 = (float*)take(256 * 4);
    float *scU2 = (float*)take(128 * 4), *shU2 = (float*)take(128 * 4);
    float *scD2 = (float*)take(128 * 4), *shD2 = (float*)take(128 * 4);
    float *scU3 = (float*)take(64 * 4),  *shU3 = (float*)take(64 * 4);
    float *scD3 = (float*)take(64 * 4),  *shD3 = (float*)take(64 * 4);
    half_t* buf0 = (half_t*)take(sizeof(half_t) * (size_t)2 * 256 * 256 * 128); // 33.5 MB
    half_t* buf1 = (half_t*)take(sizeof(half_t) * (size_t)2 * 512 * 512 * 64);  // 67 MB

    auto cdiv = [](long long a, long long b) { return (int)((a + b - 1) / b); };

    // --- weight packing + BN folding + input mask/cast ---
    pack_weights<<<cdiv(9LL * 512 * 256, 256), 256, 0, stream>>>(W_up1, wfU1, 512, 256);
    pack_weights<<<cdiv(9LL * 256 * 256, 256), 256, 0, stream>>>(W_dec1, wfD1, 256, 256);
    pack_weights<<<cdiv(9LL * 256 * 128, 256), 256, 0, stream>>>(W_up2, wfU2, 256, 128);
    pack_weights<<<cdiv(9LL * 128 * 128, 256), 256, 0, stream>>>(W_dec2, wfD2, 128, 128);
    pack_weights<<<cdiv(9LL * 128 * 64, 256), 256, 0, stream>>>(W_up3, wfU3, 128, 64);
    pack_weights<<<cdiv(9LL * 64 * 64, 256), 256, 0, stream>>>(W_dec3, wfD3, 64, 64);
    bn_prep<<<1, 256, 0, stream>>>(g_up1, be_up1, mu_up1, v_up1, b_up1, scU1, shU1, 256);
    bn_prep<<<1, 256, 0, stream>>>(g_dec1, be_dec1, mu_dec1, v_dec1, nullptr, scD1, shD1, 256);
    bn_prep<<<1, 256, 0, stream>>>(g_up2, be_up2, mu_up2, v_up2, b_up2, scU2, shU2, 128);
    bn_prep<<<1, 256, 0, stream>>>(g_dec2, be_dec2, mu_dec2, v_dec2, nullptr, scD2, shD2, 128);
    bn_prep<<<1, 256, 0, stream>>>(g_up3, be_up3, mu_up3, v_up3, b_up3, scU3, shU3, 64);
    bn_prep<<<1, 256, 0, stream>>>(g_dec3, be_dec3, mu_dec3, v_dec3, nullptr, scD3, shD3, 64);
    mask_cast<<<cdiv(2LL * 64 * 64 * 512, 256), 256, 0, stream>>>(
        x, m3, buf0, 512, 2LL * 64 * 64 * 512);

    // --- up1: [2,64,64,512] -> [2,128,128,256] ---
    { const int tiles = 2 * 128 * 2 * (64 / 16) * (256 / 64);   // 8192
      inv_conv_bn_relu<512, 256><<<cdiv(tiles, 8), 256, 0, stream>>>(
          buf0, wfU1, scU1, shU1, m2, buf1, 2, 64, 64); }
    // --- dec1: subm 256->256 @128 : blocks = B*H*(WT/8)*CT ---
    { const int blocks = 2 * 128 * (128 / 16 / 8) * (256 / 64); // 1024
      subm_conv_bn_relu<256, 256><<<blocks, 256, 0, stream>>>(
          buf1, wfD1, scD1, shD1, m2, buf0, nullptr, 2, 128, 128); }
    // --- up2: [2,128,128,256] -> [2,256,256,128] ---
    { const int tiles = 2 * 256 * 2 * (128 / 16) * (128 / 64);  // 16384
      inv_conv_bn_relu<256, 128><<<cdiv(tiles, 8), 256, 0, stream>>>(
          buf0, wfU2, scU2, shU2, m1, buf1, 2, 128, 128); }
    // --- dec2: subm 128->128 @256 ---
    { const int blocks = 2 * 256 * (256 / 16 / 8) * (128 / 64); // 2048
      subm_conv_bn_relu<128, 128><<<blocks, 256, 0, stream>>>(
          buf1, wfD2, scD2, shD2, m1, buf0, nullptr, 2, 256, 256); }
    // --- up3: [2,256,256,128] -> [2,512,512,64] ---
    { const int tiles = 2 * 512 * 2 * (256 / 16) * (64 / 64);   // 32768
      inv_conv_bn_relu<128, 64><<<cdiv(tiles, 8), 256, 0, stream>>>(
          buf0, wfU3, scU3, shU3, m0, buf1, 2, 256, 256); }
    // --- dec3: subm 64->64 @512 -> f32 d_out ---
    { const int blocks = 2 * 512 * (512 / 16 / 8) * (64 / 64);  // 4096
      subm_conv_bn_relu<64, 64><<<blocks, 256, 0, stream>>>(
          buf1, wfD3, scD3, shD3, m0, nullptr, (float*)d_out, 2, 512, 512); }
}
